// PointNet_Estimation_90812788507211
// MI455X (gfx1250) — matrix operand order repaired and checked by reading, not compiled, not run on hardware
//
#include <hip/hip_runtime.h>
#include <hip/hip_bf16.h>

// ---------------------------------------------------------------------------
// PointNet++ (SSG) forward for MI455X / gfx1250, wave32 + WMMA f16 + async LDS.
// Input order assumed = setup_inputs() insertion order:
//   d_in[0]  xyz [8,8192,3] f32
//   d_in[1..12]  sa1: 3 layers x (W,b,s,c)
//   d_in[13..24] sa2: 3 layers x (W,b,s,c)
//   d_in[25..36] sa3: 3 layers x (W,b,s,c)
//   d_in[37] fc1.W  d_in[38] fc1.b  d_in[39] bn.s  d_in[40] bn.c
//   d_in[41] fc2.W  d_in[42] fc2.b
// ---------------------------------------------------------------------------

typedef __attribute__((ext_vector_type(16))) _Float16 v16h;
typedef __attribute__((ext_vector_type(8)))  float    v8f;
typedef int v4i __attribute__((vector_size(4 * sizeof(int))));

#define WMMA16(a, b, c) \
  __builtin_amdgcn_wmma_f32_16x16x32_f16(false, (a), false, (b), (short)0, (c), false, false)

// CDNA5 async global->LDS copy (16B), tracked by ASYNCcnt.
__device__ __forceinline__ void async_copy_b128(const void* g, void* l) {
  __builtin_amdgcn_global_load_async_to_lds_b128((v4i*)g, (v4i*)l, 0, 0);
}
__device__ __forceinline__ void wait_async0() {
  asm volatile("s_wait_asynccnt 0x0" ::: "memory");
}

// ---- WMMA fragment helpers -------------------------------------------------
// A (16x32 f16, MxK) from LDS row-major [rows][ldk]:
//   lane m=lane&15, half=lane>>4; VGPR v holds K pair at
//   basek = (v&4?16:0) + (half?8:0) + (v&3)*2
__device__ __forceinline__ v16h load_afrag(const _Float16* As, int ldk, int m0, int lane) {
  const int m   = m0 + (lane & 15);
  const int khi = (lane >> 4) ? 8 : 0;
  const _Float16* row = As + (size_t)m * ldk;
  v16h a;
#pragma unroll
  for (int v = 0; v < 8; ++v) {
    const int k = ((v & 4) ? 16 : 0) + khi + ((v & 3) << 1);
    a[2 * v]     = row[k];
    a[2 * v + 1] = row[k + 1];
  }
  return a;
}

// B (32x16 f16, KxN) from pre-swizzled global weights:
//   tile (kc,nt) is 512 contiguous halves; lane-major, 16 halves (32B) per lane.
__device__ __forceinline__ v16h load_bfrag(const _Float16* W, int ntn, int kc, int nt, int lane) {
  const _Float16* p = W + (((size_t)(kc * ntn + nt) * 32) + lane) * 16;
  return *(const v16h*)p;   // -> 2x global_load_b128
}

// C epilogue relu((x+b)*s+c) -> f16 activation buffer, row-major [.. ][ldn]
__device__ __forceinline__ void store_act(v8f acc, _Float16* out, int ldn, int m0, int n0,
                                          const float* bias, const float* bs, const float* bc,
                                          int lane) {
  const int col = lane & 15;
  const int rb  = m0 + ((lane >> 4) << 3);
  const float bb = bias[n0 + col], ssv = bs[n0 + col], ccv = bc[n0 + col];
#pragma unroll
  for (int i = 0; i < 8; ++i) {
    float y = (acc[i] + bb) * ssv + ccv;
    y = y > 0.f ? y : 0.f;
    out[(size_t)(rb + i) * ldn + n0 + col] = (_Float16)y;
  }
}

// ---- weight f32 -> swizzled f16 B-tile layout (zero pad K to Kpad) ---------
// rot131: SA2 L1 input rows reordered as [feats(128) | gxyz(3)] so the async
// feature gather lands 16B-aligned in LDS; remap W rows to match.
__global__ __launch_bounds__(256) void swizzle_w_kernel(const float* __restrict__ src,
                                                        _Float16* __restrict__ dst,
                                                        int Korig, int Kpad, int N, int rot131) {
  const int total = Kpad * N;
  const int ntn = N >> 4;
  for (int e = blockIdx.x * 256 + threadIdx.x; e < total; e += gridDim.x * 256) {
    const int h    = e & 15;
    const int lane = (e >> 4) & 31;
    const int blk  = e >> 9;
    const int nt = blk % ntn, kc = blk / ntn;
    const int col = nt * 16 + (lane & 15);
    const int k   = kc * 32 + ((h & 8) << 1) + ((lane >> 4) << 3) + (h & 7);
    int ks = k;
    if (rot131) ks = (k < 128) ? (k + 3) : (k - 128);   // cols 0..127 = W rows 3..130
    dst[e] = (k < Korig) ? (_Float16)src[(size_t)ks * N + col] : (_Float16)0.f;
  }
}

// ---- farthest point sampling (one block per batch) -------------------------
__global__ __launch_bounds__(256) void fps_kernel(const float* __restrict__ xyz, int N, int Kout,
                                                  int* __restrict__ out_idx,
                                                  float* __restrict__ out_xyz) {
  __shared__ float dist[8192];
  __shared__ float rmax[256];
  __shared__ int   rid[256];
  __shared__ int   cur;
  const int b = blockIdx.x;
  const float* px = xyz + (size_t)b * N * 3;
  for (int i = threadIdx.x; i < N; i += 256) dist[i] = 3.4e38f;
  if (threadIdx.x == 0) cur = 0;
  __syncthreads();
  for (int it = 0; it < Kout; ++it) {
    const int c = cur;
    const float cx = px[c * 3], cy = px[c * 3 + 1], cz = px[c * 3 + 2];
    if (threadIdx.x == 0) {
      out_idx[b * Kout + it] = c;
      out_xyz[(size_t)(b * Kout + it) * 3 + 0] = cx;
      out_xyz[(size_t)(b * Kout + it) * 3 + 1] = cy;
      out_xyz[(size_t)(b * Kout + it) * 3 + 2] = cz;
    }
    float lm = -1.f; int li = 0;
    for (int i = threadIdx.x; i < N; i += 256) {
      const float dx = px[i * 3] - cx, dy = px[i * 3 + 1] - cy, dz = px[i * 3 + 2] - cz;
      const float nd = fminf(dist[i], dx * dx + dy * dy + dz * dz);
      dist[i] = nd;
      if (nd > lm) { lm = nd; li = i; }
    }
    rmax[threadIdx.x] = lm; rid[threadIdx.x] = li;
    __syncthreads();
    for (int s = 128; s > 0; s >>= 1) {
      if (threadIdx.x < (unsigned)s && rmax[threadIdx.x + s] > rmax[threadIdx.x]) {
        rmax[threadIdx.x] = rmax[threadIdx.x + s];
        rid[threadIdx.x]  = rid[threadIdx.x + s];
      }
      __syncthreads();
    }
    if (threadIdx.x == 0) cur = rid[0];
    __syncthreads();
  }
}

// ---- SA1: ball query + gather + MLP(3->64->64->128) + maxpool(K=32) --------
// one group per wave; 4 waves / block; weights pre-swizzled in global (L2-hot)
__global__ __launch_bounds__(128) void sa1_kernel(
    const float* __restrict__ xyz, const float* __restrict__ nxyz,
    const _Float16* __restrict__ W1, const float* b1, const float* s1, const float* c1,
    const _Float16* __restrict__ W2, const float* b2, const float* s2, const float* c2,
    const _Float16* __restrict__ W3, const float* b3, const float* s3, const float* c3,
    _Float16* __restrict__ l1pts) {
  __shared__ _Float16 buf0[4][32 * 64];
  __shared__ _Float16 buf1[4][32 * 64];
  __shared__ int nidx[4][32];
  const int lane = threadIdx.x & 31, wave = threadIdx.x >> 5;
  const int g = blockIdx.x * 4 + wave;          // 0..4095
  const int b = g >> 9;
  const float* ctr = nxyz + (size_t)g * 3;
  const float cx = ctr[0], cy = ctr[1], cz = ctr[2];
  const float* bx = xyz + (size_t)b * 8192 * 3;

  // --- ball query (first-32 by index within r=0.2), pad with N-1 ---
  nidx[wave][lane] = 8191;
  int cnt = 0;
  for (int base = 0; base < 8192 && cnt < 32; base += 32) {
    const float* q = bx + (size_t)(base + lane) * 3;
    const float dx = q[0] - cx, dy = q[1] - cy, dz = q[2] - cz;
    const bool win = (dx * dx + dy * dy + dz * dz) < 0.04f;
    const unsigned mask = (unsigned)__ballot(win);
    if (win) {
      const int rank = cnt + __popc(mask & ((1u << lane) - 1u));
      if (rank < 32) nidx[wave][rank] = base + lane;
    }
    cnt += __popc(mask);
  }
  // --- gather A0 [32 x 32(pad from 3)] into buf1 ---
  {
    const int idx = nidx[wave][lane];
    const float* q = bx + (size_t)idx * 3;
    _Float16* row = &buf1[wave][lane * 32];
    row[0] = (_Float16)(q[0] - cx);
    row[1] = (_Float16)(q[1] - cy);
    row[2] = (_Float16)(q[2] - cz);
#pragma unroll
    for (int k = 3; k < 32; ++k) row[k] = (_Float16)0.f;
  }
  // --- L1: [32x32]@[32x64] -> buf0 ---
#pragma unroll
  for (int nt = 0; nt < 4; ++nt) {
    const v16h bf = load_bfrag(W1, 4, 0, nt, lane);
#pragma unroll
    for (int rt = 0; rt < 2; ++rt) {
      const v16h a = load_afrag(&buf1[wave][0], 32, rt * 16, lane);
      v8f acc = {};
      acc = WMMA16(a, bf, acc);
      store_act(acc, &buf0[wave][0], 64, rt * 16, nt * 16, b1, s1, c1, lane);
    }
  }
  // --- L2: [32x64]@[64x64] -> buf1 ---
#pragma unroll
  for (int nt = 0; nt < 4; ++nt) {
    const v16h bf0 = load_bfrag(W2, 4, 0, nt, lane);
    const v16h bf1 = load_bfrag(W2, 4, 1, nt, lane);
#pragma unroll
    for (int rt = 0; rt < 2; ++rt) {
      v8f acc = {};
      v16h a = load_afrag(&buf0[wave][0], 64, rt * 16, lane);
      acc = WMMA16(a, bf0, acc);
      a = load_afrag(&buf0[wave][0] + 32, 64, rt * 16, lane);
      acc = WMMA16(a, bf1, acc);
      store_act(acc, &buf1[wave][0], 64, rt * 16, nt * 16, b2, s2, c2, lane);
    }
  }
  // --- L3: [32x64]@[64x128] + relu + max over 32 rows -> l1pts ---
#pragma unroll
  for (int nt = 0; nt < 8; ++nt) {
    const v16h bf0 = load_bfrag(W3, 8, 0, nt, lane);
    const v16h bf1 = load_bfrag(W3, 8, 1, nt, lane);
    const int col = lane & 15;
    const float bb = b3[nt * 16 + col], ssv = s3[nt * 16 + col], ccv = c3[nt * 16 + col];
    float cm = -3.4e38f;
#pragma unroll
    for (int rt = 0; rt < 2; ++rt) {
      v8f acc = {};
      v16h a = load_afrag(&buf1[wave][0], 64, rt * 16, lane);
      acc = WMMA16(a, bf0, acc);
      a = load_afrag(&buf1[wave][0] + 32, 64, rt * 16, lane);
      acc = WMMA16(a, bf1, acc);
#pragma unroll
      for (int i = 0; i < 8; ++i) {
        float y = (acc[i] + bb) * ssv + ccv;
        y = y > 0.f ? y : 0.f;
        cm = fmaxf(cm, y);
      }
    }
    cm = fmaxf(cm, __shfl_xor(cm, 16, 32));
    if (lane < 16) l1pts[(size_t)g * 128 + nt * 16 + lane] = (_Float16)cm;
  }
}

// ---- SA2: ball query(K=64,r=0.4) + async gather(128+3) +
//      MLP(160->128->128->256) + maxpool(64); writes padded SA3 row [288].
//      2 waves per group. Input row layout: [feats(128) | gxyz(3) | 0(29)].
__global__ __launch_bounds__(64) void sa2_kernel(
    const float* __restrict__ xyz1, const float* __restrict__ xyz2,
    const _Float16* __restrict__ l1pts,
    const _Float16* __restrict__ W1, const float* b1, const float* s1, const float* c1,
    const _Float16* __restrict__ W2, const float* b2, const float* s2, const float* c2,
    const _Float16* __restrict__ W3, const float* b3, const float* s3, const float* c3,
    _Float16* __restrict__ A3) {
  __shared__ _Float16 A0[64 * 160];
  __shared__ _Float16 buf1[64 * 128];
  __shared__ float pmax[2][256];
  __shared__ int nidx[64];
  const int tid = threadIdx.x, lane = tid & 31, wave = tid >> 5;
  const int g = blockIdx.x;                      // 0..1023
  const int b = g >> 7;
  const float* ctr = xyz2 + (size_t)g * 3;
  const float cx = ctr[0], cy = ctr[1], cz = ctr[2];
  const float* bx = xyz1 + (size_t)b * 512 * 3;

  if (wave == 0) {
    nidx[lane] = 511; nidx[32 + lane] = 511;
    int cnt = 0;
    for (int base = 0; base < 512 && cnt < 64; base += 32) {
      const float* q = bx + (size_t)(base + lane) * 3;
      const float dx = q[0] - cx, dy = q[1] - cy, dz = q[2] - cz;
      const bool win = (dx * dx + dy * dy + dz * dz) < 0.16f;
      const unsigned mask = (unsigned)__ballot(win);
      if (win) {
        const int rank = cnt + __popc(mask & ((1u << lane) - 1u));
        if (rank < 64) nidx[rank] = base + lane;
      }
      cnt += __popc(mask);
    }
  }
  __syncthreads();
  // gather row tid: feats via CDNA5 async DMA to LDS (16B aligned), xyz tail VALU
  {
    const int idx = nidx[tid];
    _Float16* row = A0 + (size_t)tid * 160;
    const _Float16* f = l1pts + ((size_t)b * 512 + idx) * 128;
#pragma unroll
    for (int j = 0; j < 16; ++j)
      async_copy_b128(f + j * 8, row + j * 8);    // 256B feature block
    const float* q = bx + (size_t)idx * 3;
    row[128] = (_Float16)(q[0] - cx);
    row[129] = (_Float16)(q[1] - cy);
    row[130] = (_Float16)(q[2] - cz);
#pragma unroll
    for (int j = 131; j < 160; ++j) row[j] = (_Float16)0.f;
  }
  wait_async0();
  __syncthreads();
  // L1: [64x160]@[160x128] -> buf1 (wave handles row-tiles 2w,2w+1)
  for (int nt = 0; nt < 8; ++nt) {
#pragma unroll
    for (int rt2 = 0; rt2 < 2; ++rt2) {
      const int rt = wave * 2 + rt2;
      v8f acc = {};
#pragma unroll
      for (int kc = 0; kc < 5; ++kc) {
        const v16h a  = load_afrag(A0 + kc * 32, 160, rt * 16, lane);
        const v16h bf = load_bfrag(W1, 8, kc, nt, lane);
        acc = WMMA16(a, bf, acc);
      }
      store_act(acc, buf1, 128, rt * 16, nt * 16, b1, s1, c1, lane);
    }
  }
  __syncthreads();
  _Float16* buf2 = A0;  // A0 dead; reuse as [64x128]
  // L2: [64x128]@[128x128] -> buf2
  for (int nt = 0; nt < 8; ++nt) {
#pragma unroll
    for (int rt2 = 0; rt2 < 2; ++rt2) {
      const int rt = wave * 2 + rt2;
      v8f acc = {};
#pragma unroll
      for (int kc = 0; kc < 4; ++kc) {
        const v16h a  = load_afrag(buf1 + kc * 32, 128, rt * 16, lane);
        const v16h bf = load_bfrag(W2, 8, kc, nt, lane);
        acc = WMMA16(a, bf, acc);
      }
      store_act(acc, buf2, 128, rt * 16, nt * 16, b2, s2, c2, lane);
    }
  }
  __syncthreads();
  // L3: [64x128]@[128x256] + relu + per-wave max over its 32 rows
  for (int nt = 0; nt < 16; ++nt) {
    const int col = lane & 15;
    const float bb = b3[nt * 16 + col], ssv = s3[nt * 16 + col], ccv = c3[nt * 16 + col];
    float cm = -3.4e38f;
#pragma unroll
    for (int rt2 = 0; rt2 < 2; ++rt2) {
      const int rt = wave * 2 + rt2;
      v8f acc = {};
#pragma unroll
      for (int kc = 0; kc < 4; ++kc) {
        const v16h a  = load_afrag(buf2 + kc * 32, 128, rt * 16, lane);
        const v16h bf = load_bfrag(W3, 16, kc, nt, lane);
        acc = WMMA16(a, bf, acc);
      }
#pragma unroll
      for (int i = 0; i < 8; ++i) {
        float y = (acc[i] + bb) * ssv + ccv;
        y = y > 0.f ? y : 0.f;
        cm = fmaxf(cm, y);
      }
    }
    cm = fmaxf(cm, __shfl_xor(cm, 16, 32));
    if (lane < 16) pmax[wave][nt * 16 + lane] = cm;
  }
  __syncthreads();
  // combine waves + emit padded SA3 row: [center(3) | feats(256) | 0(29)]
  _Float16* arow = A3 + (size_t)g * 288;
  for (int j = tid; j < 256; j += 64)
    arow[3 + j] = (_Float16)fmaxf(pmax[0][j], pmax[1][j]);
  if (tid == 0) { arow[0] = (_Float16)cx; arow[1] = (_Float16)cy; arow[2] = (_Float16)cz; }
  for (int j = 259 + tid; j < 288; j += 64) arow[j] = (_Float16)0.f;
}

// ---- generic WMMA GEMM: out = [relu]((A@W + b)[*s + c]), 128x64 block tile -
// A tile staged LDS via CDNA5 async global->LDS DMA.
__global__ __launch_bounds__(256) void gemm_bn_kernel(
    const _Float16* __restrict__ A, const _Float16* __restrict__ W,
    const float* __restrict__ bias, const float* __restrict__ bs, const float* __restrict__ bc,
    _Float16* __restrict__ out16, float* __restrict__ out32,
    int M, int K, int N, int relu) {
  __shared__ _Float16 As[128 * 32];
  const int lane = threadIdx.x & 31, wave = threadIdx.x >> 5;
  const int row0 = blockIdx.y * 128;
  const int col0 = blockIdx.x * 64;
  const int ntn = N >> 4;
  const int m0 = wave * 16;
  const bool active = (row0 + m0) < M;
  v8f acc[4] = {{}, {}, {}, {}};
  const int nkc = K >> 5;
  for (int kc = 0; kc < nkc; ++kc) {
    const int r  = threadIdx.x >> 1;
    const int cp = (threadIdx.x & 1) << 4;
    _Float16* dp = As + r * 32 + cp;
    if (row0 + r < M) {
      const _Float16* sp = A + (size_t)(row0 + r) * K + kc * 32 + cp;
      async_copy_b128(sp, dp);
      async_copy_b128(sp + 8, dp + 8);
    } else {
      uint4 z = {0u, 0u, 0u, 0u};
      uint4* d = (uint4*)dp;
      d[0] = z; d[1] = z;
    }
    wait_async0();
    __syncthreads();
    if (active) {
      const v16h a = load_afrag(As, 32, m0, lane);
#pragma unroll
      for (int nt = 0; nt < 4; ++nt) {
        const v16h bf = load_bfrag(W, ntn, kc, (col0 >> 4) + nt, lane);
        acc[nt] = WMMA16(a, bf, acc[nt]);
      }
    }
    __syncthreads();
  }
  if (active) {
    const int col = lane & 15;
    const int rb  = row0 + m0 + ((lane >> 4) << 3);
#pragma unroll
    for (int nt = 0; nt < 4; ++nt) {
      const int gcol = col0 + nt * 16 + col;
      const float bb  = bias ? bias[gcol] : 0.f;
      const float ssv = bs ? bs[gcol] : 1.f;
      const float ccv = bs ? bc[gcol] : 0.f;
#pragma unroll
      for (int i = 0; i < 8; ++i) {
        const int grow = rb + i;
        if (grow < M) {
          float y = (acc[nt][i] + bb) * ssv + ccv;
          if (relu) y = y > 0.f ? y : 0.f;
          if (out16) out16[(size_t)grow * N + gcol] = (_Float16)y;
          if (out32) out32[(size_t)grow * N + gcol] = y;
        }
      }
    }
  }
}

// ---- SA3 global max over 128 points -> [16,1024] f16 (rows 8..15 zero) -----
__global__ __launch_bounds__(256) void maxpool3_kernel(const _Float16* __restrict__ h,
                                                       _Float16* __restrict__ gmax) {
  const int b = blockIdx.x;
  const int ch = blockIdx.y * 256 + threadIdx.x;
  float m = -3.4e38f;
  for (int r = 0; r < 128; ++r)
    m = fmaxf(m, (float)h[((size_t)(b * 128 + r)) * 1024 + ch]);
  gmax[(size_t)b * 1024 + ch] = (_Float16)m;
  gmax[(size_t)(8 + b) * 1024 + ch] = (_Float16)0.f;
}

// ---- log_softmax over 512 classes, one block per batch row -----------------
__global__ __launch_bounds__(512) void logsoftmax_kernel(const float* __restrict__ in,
                                                         float* __restrict__ out) {
  __shared__ float red[512];
  const int r = blockIdx.x, t = threadIdx.x;
  const float x = in[(size_t)r * 512 + t];
  red[t] = x;
  __syncthreads();
  for (int s = 256; s > 0; s >>= 1) {
    if (t < s) red[t] = fmaxf(red[t], red[t + s]);
    __syncthreads();
  }
  const float mx = red[0];
  __syncthreads();
  red[t] = expf(x - mx);
  __syncthreads();
  for (int s = 256; s > 0; s >>= 1) {
    if (t < s) red[t] += red[t + s];
    __syncthreads();
  }
  out[(size_t)r * 512 + t] = x - (logf(red[0]) + mx);
}

// ---------------------------------------------------------------------------
extern "C" void kernel_launch(void* const* d_in, const int* in_sizes, int n_in,
                              void* d_out, int out_size, void* d_ws, size_t ws_size,
                              hipStream_t stream) {
  (void)in_sizes; (void)n_in; (void)out_size; (void)ws_size;
  const float* xyz = (const float*)d_in[0];
  const float *saW[9], *saB[9], *saS[9], *saC[9];
  int ii = 1;
  for (int l = 0; l < 9; ++l) {
    saW[l] = (const float*)d_in[ii++];
    saB[l] = (const float*)d_in[ii++];
    saS[l] = (const float*)d_in[ii++];
    saC[l] = (const float*)d_in[ii++];
  }
  const float* fc1W = (const float*)d_in[37];
  const float* fc1b = (const float*)d_in[38];
  const float* bns  = (const float*)d_in[39];
  const float* bnc  = (const float*)d_in[40];
  const float* fc2W = (const float*)d_in[41];
  const float* fc2b = (const float*)d_in[42];

  char* wsb = (char*)d_ws;
  size_t off = 0;
  auto carve = [&](size_t bytes) -> void* {
    void* p = wsb + off;
    off = (off + bytes + 255) & ~(size_t)255;
    return p;
  };
  int*       fps1_idx = (int*)carve(8 * 512 * 4);
  float*     xyz1     = (float*)carve((size_t)8 * 512 * 3 * 4);
  _Float16*  l1pts    = (_Float16*)carve((size_t)8 * 512 * 128 * 2);
  int*       fps2_idx = (int*)carve(8 * 128 * 4);
  float*     xyz2     = (float*)carve((size_t)8 * 128 * 3 * 4);
  _Float16*  A3       = (_Float16*)carve((size_t)1024 * 288 * 2);
  _Float16*  h3a      = (_Float16*)carve((size_t)1024 * 256 * 2);
  _Float16*  h3b      = (_Float16*)carve((size_t)1024 * 512 * 2);
  _Float16*  h3c      = (_Float16*)carve((size_t)1024 * 1024 * 2);
  _Float16*  gmax     = (_Float16*)carve((size_t)16 * 1024 * 2);
  _Float16*  fc1o     = (_Float16*)carve((size_t)16 * 1024 * 2);
  float*     fc2o     = (float*)carve((size_t)16 * 512 * 4);

  // swizzled weights: {src, Korig, Kpad, N, rot131}
  struct WS { const float* src; int Ko, Kp, N, rot; _Float16* dst; };
  WS wz[11] = {
      {saW[0],   3,   32,   64, 0, nullptr},  // sa1 L1
      {saW[1],  64,   64,   64, 0, nullptr},  // sa1 L2
      {saW[2],  64,   64,  128, 0, nullptr},  // sa1 L3
      {saW[3], 131,  160,  128, 1, nullptr},  // sa2 L1 (row-rotated for async gather)
      {saW[4], 128,  128,  128, 0, nullptr},  // sa2 L2
      {saW[5], 128,  128,  256, 0, nullptr},  // sa2 L3
      {saW[6], 259,  288,  256, 0, nullptr},  // sa3 L1
      {saW[7], 256,  256,  512, 0, nullptr},  // sa3 L2
      {saW[8], 512,  512, 1024, 0, nullptr},  // sa3 L3
      {fc1W, 1024, 1024, 1024, 0, nullptr},
      {fc2W, 1024, 1024,  512, 0, nullptr},
  };
  for (int i = 0; i < 11; ++i) {
    wz[i].dst = (_Float16*)carve((size_t)wz[i].Kp * wz[i].N * 2);
    const int total = wz[i].Kp * wz[i].N;
    int grid = (total + 255) / 256;
    if (grid > 2048) grid = 2048;
    swizzle_w_kernel<<<grid, 256, 0, stream>>>(wz[i].src, wz[i].dst,
                                               wz[i].Ko, wz[i].Kp, wz[i].N, wz[i].rot);
  }

  // stage 1
  fps_kernel<<<8, 256, 0, stream>>>(xyz, 8192, 512, fps1_idx, xyz1);
  sa1_kernel<<<1024, 128, 0, stream>>>(xyz, xyz1,
      wz[0].dst, saB[0], saS[0], saC[0],
      wz[1].dst, saB[1], saS[1], saC[1],
      wz[2].dst, saB[2], saS[2], saC[2], l1pts);

  // stage 2
  fps_kernel<<<8, 256, 0, stream>>>(xyz1, 512, 128, fps2_idx, xyz2);
  sa2_kernel<<<1024, 64, 0, stream>>>(xyz1, xyz2, l1pts,
      wz[3].dst, saB[3], saS[3], saC[3],
      wz[4].dst, saB[4], saS[4], saC[4],
      wz[5].dst, saB[5], saS[5], saC[5], A3);

  // stage 3 (group_all MLP as GEMMs over 1024 rows)
  gemm_bn_kernel<<<dim3(4, 8), 256, 0, stream>>>(A3,  wz[6].dst, saB[6], saS[6], saC[6],
                                                 h3a, nullptr, 1024, 288, 256, 1);
  gemm_bn_kernel<<<dim3(8, 8), 256, 0, stream>>>(h3a, wz[7].dst, saB[7], saS[7], saC[7],
                                                 h3b, nullptr, 1024, 256, 512, 1);
  gemm_bn_kernel<<<dim3(16, 8), 256, 0, stream>>>(h3b, wz[8].dst, saB[8], saS[8], saC[8],
                                                  h3c, nullptr, 1024, 512, 1024, 1);
  maxpool3_kernel<<<dim3(8, 4), 256, 0, stream>>>(h3c, gmax);

  // classifier head
  gemm_bn_kernel<<<dim3(16, 1), 256, 0, stream>>>(gmax, wz[9].dst, fc1b, bns, bnc,
                                                  fc1o, nullptr, 16, 1024, 1024, 1);
  gemm_bn_kernel<<<dim3(8, 1), 256, 0, stream>>>(fc1o, wz[10].dst, fc2b, nullptr, nullptr,
                                                 nullptr, fc2o, 16, 1024, 512, 0);
  logsoftmax_kernel<<<8, 512, 0, stream>>>(fc2o, (float*)d_out);
}
